// DecoupledFlowMatching_49658411876460
// MI455X (gfx1250) — matrix operand order repaired and checked by reading, not compiled, not run on hardware
//
#include <hip/hip_runtime.h>
#include <hip/hip_bf16.h>

// MI455X / gfx1250, wave32. f16 WMMA (v_wmma_f32_16x16x32_f16) for all GEMMs.
// B=65536, D=64, H=E=1024. Weights pre-packed once to f16 [N][K] so B-staging
// is pure b128 copies via GLOBAL_LOAD_ASYNC_TO_LDS (ASYNCcnt-tracked).

#define BB 65536
#define DD 64
#define HH 1024

typedef __attribute__((ext_vector_type(16))) _Float16 v16h;
typedef __attribute__((ext_vector_type(8)))  float    v8f;
typedef __attribute__((ext_vector_type(4)))  int      v4i;

#if defined(__HIP_DEVICE_COMPILE__) && \
    __has_builtin(__builtin_amdgcn_global_load_async_to_lds_b128) && \
    __has_builtin(__builtin_amdgcn_s_wait_asynccnt)
#define ASYNC_LDS 1
#endif

// 16-byte global -> LDS copy (async-to-LDS when available)
__device__ __forceinline__ void cp16(_Float16* l, const _Float16* g) {
#ifdef ASYNC_LDS
    __builtin_amdgcn_global_load_async_to_lds_b128(
        (__attribute__((address_space(1))) v4i*)g,
        (__attribute__((address_space(3))) v4i*)l,
        0, 0);
#else
    *(uint4*)l = *(const uint4*)g;
#endif
}

__device__ __forceinline__ void cp_commit() {
#ifdef ASYNC_LDS
    __builtin_amdgcn_s_wait_asynccnt(0);
#endif
}

__device__ __forceinline__ float silu_f(float x) {
    return x / (1.0f + __expf(-x));
}

// ---- fragment loaders -------------------------------------------------------
// A (16x32 f16 in LDS As[16][32]):
//   lane<16 : M=lane,    VGPR0-3 = K0..7,  VGPR4-7 = K16..23
//   lane>=16: M=lane-16, VGPR0-3 = K8..15, VGPR4-7 = K24..31
__device__ __forceinline__ v16h load_a_frag(const _Float16* As, int lane) {
    int m  = lane & 15;
    int k0 = (lane < 16) ? 0 : 8;
    union { uint4 u[2]; v16h h; } f;
    f.u[0] = *(const uint4*)(As + m * 32 + k0);
    f.u[1] = *(const uint4*)(As + m * 32 + k0 + 16);
    return f.h;
}

// B (32x16 f16) staged as Bs[col][32]: lane<16 => col=nb+lane, K=0..7,16..23;
// lane>=16 => K=8..15,24..31. 2 x b128 per lane.
__device__ __forceinline__ v16h load_b_frag(const _Float16* Bs, int nb, int lane) {
    int c  = nb + (lane & 15);
    int k0 = (lane < 16) ? 0 : 8;
    union { uint4 u[2]; v16h h; } f;
    f.u[0] = *(const uint4*)(Bs + c * 32 + k0);
    f.u[1] = *(const uint4*)(Bs + c * 32 + k0 + 16);
    return f.h;
}

// ---- kernel 0: pack weight f32 [K][N] -> f16 transposed [N][K] --------------
__global__ __launch_bounds__(256) void k_packT(const float* __restrict__ src,
                                               _Float16* __restrict__ dst,
                                               int K, int N) {
    int i = blockIdx.x * 256 + threadIdx.x;   // over K*N, coalesced read
    int k = i / N;
    int n = i - k * N;
    dst[(size_t)n * K + k] = (_Float16)src[i];
}

// ---- kernel 1: mixed = t*gt + (1-t)*noise  (f32 -> f16) ---------------------
__global__ __launch_bounds__(256) void k_mixed(const float* __restrict__ gt,
                                               const float* __restrict__ noise,
                                               const float* __restrict__ t,
                                               _Float16* __restrict__ mixed) {
    int i = blockIdx.x * 256 + threadIdx.x;   // over B*D
    float tv = t[i >> 6];
    mixed[i] = (_Float16)(tv * gt[i] + (1.0f - tv) * noise[i]);
}

// ---- kernel 2: te = silu( silu(t*Wt1+bt1) @ Wt2 + bt2 ) ---------------------
// One WG = 8 waves = 16 rows x 1024 cols stripe. u computed on the fly.
__global__ __launch_bounds__(256) void k_te(const float* __restrict__ t,
                                            const float* __restrict__ Wt1,
                                            const float* __restrict__ bt1,
                                            const _Float16* __restrict__ Wt2T, // [1024][1024]
                                            const float* __restrict__ bt2,
                                            _Float16* __restrict__ te) {
    extern __shared__ char smem[];
    _Float16* As = (_Float16*)smem;            // 16*32 halves
    _Float16* Bs = (_Float16*)(smem + 1024);   // 1024*32 halves (64 KB)

    const int tid  = threadIdx.x;
    const int wave = tid >> 5;
    const int lane = tid & 31;
    const int row0 = blockIdx.x * 16;

    const v8f vzero = {0.f, 0.f, 0.f, 0.f, 0.f, 0.f, 0.f, 0.f};
    v8f acc[8];
#pragma unroll
    for (int i = 0; i < 8; ++i) acc[i] = vzero;

    for (int kc = 0; kc < 1024; kc += 32) {
        // stage A: u = silu(t*Wt1 + bt1) for this 16x32 chunk
        for (int i = tid; i < 512; i += 256) {
            int r = i >> 5, k = i & 31;
            float tv = t[row0 + r];
            As[r * 32 + k] = (_Float16)silu_f(tv * Wt1[kc + k] + bt1[kc + k]);
        }
        // stage B: packed f16, pure b128 async copies
        for (int i = tid; i < 4096; i += 256) {
            int n = i >> 2, seg = (i & 3) << 3;
            cp16(Bs + n * 32 + seg, Wt2T + (size_t)n * 1024 + kc + seg);
        }
        if (kc + 32 < 1024)
            __builtin_prefetch(Wt2T + (size_t)tid * 1024 + kc + 32, 0, 1);
        cp_commit();
        __syncthreads();

        v16h a = load_a_frag(As, lane);
#pragma unroll
        for (int nt = 0; nt < 8; ++nt) {
            v16h b = load_b_frag(Bs, wave * 128 + nt * 16, lane);
            acc[nt] = __builtin_amdgcn_wmma_f32_16x16x32_f16(
                false, a, false, b, (short)0, acc[nt], false, false);
        }
        __syncthreads();
    }

    // epilogue: silu(acc + bt2) -> te (f16). C layout: VGPR r -> M=r+(lane/16)*8
    const int mh = (lane >> 4) * 8;
#pragma unroll
    for (int nt = 0; nt < 8; ++nt) {
        int n = wave * 128 + nt * 16 + (lane & 15);
        float bb = bt2[n];
#pragma unroll
        for (int r = 0; r < 8; ++r) {
            float v = silu_f(acc[nt][r] + bb);
            te[(size_t)(row0 + mh + r) * 1024 + n] = (_Float16)v;
        }
    }
}

// ---- kernel 3: fused block --------------------------------------------------
// h = x@W + b ; ss = te@Ws + bs ; out = silu( (LN(h)*g+be)*(1+scale)+shift )
// WG stripe: 16 rows x 1024 cols. Phase A: scale/shift -> LDS (f32).
// Phase B: h accumulators in VGPRs, cross-wave LN reduction in LDS.
template <int K>
__global__ __launch_bounds__(256) void k_block(const _Float16* __restrict__ x,
                                               const _Float16* __restrict__ te,
                                               const _Float16* __restrict__ WT,  // [1024][K]
                                               const float* __restrict__ b,
                                               const float* __restrict__ g,
                                               const float* __restrict__ be,
                                               const _Float16* __restrict__ WsT, // [2048][1024]
                                               const float* __restrict__ bs,
                                               _Float16* __restrict__ out) {
    extern __shared__ char smem[];
    _Float16* As    = (_Float16*)smem;                   //  1 KB
    _Float16* Bs0   = (_Float16*)(smem + 1024);          // 64 KB
    _Float16* Bs1   = Bs0 + 1024 * 32;                   // 64 KB
    float*    scLDS = (float*)(Bs1 + 1024 * 32);         // 64 KB (16x1024)
    float*    shLDS = scLDS + 16 * 1024;                 // 64 KB
    float*    red   = shLDS + 16 * 1024;                 // 32 floats

    const int tid  = threadIdx.x;
    const int wave = tid >> 5;
    const int lane = tid & 31;
    const int row0 = blockIdx.x * 16;
    const int mh   = (lane >> 4) * 8;

    const v8f vzero = {0.f, 0.f, 0.f, 0.f, 0.f, 0.f, 0.f, 0.f};

    // ---------------- Phase A: scale/shift = te @ Ws + bs -------------------
    {
        v8f sc[8], sh[8];
#pragma unroll
        for (int i = 0; i < 8; ++i) { sc[i] = vzero; sh[i] = vzero; }

        for (int kc = 0; kc < 1024; kc += 32) {
            if (tid < 64) {
                int r = tid >> 2, seg = (tid & 3) << 3;
                cp16(As + r * 32 + seg,
                     te + (size_t)(row0 + r) * 1024 + kc + seg);
            }
            for (int i = tid; i < 4096; i += 256) {
                int n = i >> 2, seg = (i & 3) << 3;
                cp16(Bs0 + n * 32 + seg, WsT + (size_t)n * 1024 + kc + seg);
                cp16(Bs1 + n * 32 + seg, WsT + (size_t)(n + 1024) * 1024 + kc + seg);
            }
            if (kc + 32 < 1024)
                __builtin_prefetch(WsT + (size_t)tid * 1024 + kc + 32, 0, 1);
            cp_commit();
            __syncthreads();

            v16h a = load_a_frag(As, lane);
#pragma unroll
            for (int nt = 0; nt < 8; ++nt) {
                v16h b0 = load_b_frag(Bs0, wave * 128 + nt * 16, lane);
                sc[nt] = __builtin_amdgcn_wmma_f32_16x16x32_f16(
                    false, a, false, b0, (short)0, sc[nt], false, false);
                v16h b1 = load_b_frag(Bs1, wave * 128 + nt * 16, lane);
                sh[nt] = __builtin_amdgcn_wmma_f32_16x16x32_f16(
                    false, a, false, b1, (short)0, sh[nt], false, false);
            }
            __syncthreads();
        }
        // park scale/shift (+bias) in LDS
#pragma unroll
        for (int nt = 0; nt < 8; ++nt) {
            int n = wave * 128 + nt * 16 + (lane & 15);
            float bsc = bs[n];
            float bsh = bs[n + 1024];
#pragma unroll
            for (int r = 0; r < 8; ++r) {
                int m = mh + r;
                scLDS[m * 1024 + n] = sc[nt][r] + bsc;
                shLDS[m * 1024 + n] = sh[nt][r] + bsh;
            }
        }
        __syncthreads();
    }

    // ---------------- Phase B: h = x @ W + b --------------------------------
    v8f hc[8];
#pragma unroll
    for (int i = 0; i < 8; ++i) hc[i] = vzero;

    for (int kc = 0; kc < K; kc += 32) {
        if (tid < 64) {
            int r = tid >> 2, seg = (tid & 3) << 3;
            cp16(As + r * 32 + seg,
                 x + (size_t)(row0 + r) * K + kc + seg);
        }
        for (int i = tid; i < 4096; i += 256) {
            int n = i >> 2, seg = (i & 3) << 3;
            cp16(Bs0 + n * 32 + seg, WT + (size_t)n * K + kc + seg);
        }
        cp_commit();
        __syncthreads();

        v16h a = load_a_frag(As, lane);
#pragma unroll
        for (int nt = 0; nt < 8; ++nt) {
            v16h bfr = load_b_frag(Bs0, wave * 128 + nt * 16, lane);
            hc[nt] = __builtin_amdgcn_wmma_f32_16x16x32_f16(
                false, a, false, bfr, (short)0, hc[nt], false, false);
        }
        __syncthreads();
    }

    // bias (must precede LN statistics)
#pragma unroll
    for (int nt = 0; nt < 8; ++nt) {
        int n = wave * 128 + nt * 16 + (lane & 15);
        float bb = b[n];
#pragma unroll
        for (int r = 0; r < 8; ++r) hc[nt][r] += bb;
    }

    // LN stats: lanes 0-15 own rows 0..7, lanes 16-31 rows 8..15.
    if (tid < 32) red[tid] = 0.0f;
    __syncthreads();
#pragma unroll
    for (int r = 0; r < 8; ++r) {
        float s = 0.f, q = 0.f;
#pragma unroll
        for (int nt = 0; nt < 8; ++nt) {
            float v = hc[nt][r];
            s += v;
            q += v * v;
        }
        for (int off = 8; off; off >>= 1) {
            s += __shfl_xor(s, off, 16);
            q += __shfl_xor(q, off, 16);
        }
        if ((lane & 15) == 0) {
            int m = mh + r;
            atomicAdd(&red[m], s);
            atomicAdd(&red[16 + m], q);
        }
    }
    __syncthreads();

    // LN + modulate + silu -> out (f16)
#pragma unroll
    for (int nt = 0; nt < 8; ++nt) {
        int n = wave * 128 + nt * 16 + (lane & 15);
        float gn  = g[n];
        float ben = be[n];
#pragma unroll
        for (int r = 0; r < 8; ++r) {
            int m = mh + r;
            float mu  = red[m] * (1.0f / 1024.0f);
            float var = red[16 + m] * (1.0f / 1024.0f) - mu * mu;
            float rs  = rsqrtf(var + 1e-5f);
            float y = (hc[nt][r] - mu) * rs * gn + ben;
            y = y * (1.0f + scLDS[m * 1024 + n]) + shLDS[m * 1024 + n];
            y = silu_f(y);
            out[(size_t)(row0 + m) * 1024 + n] = (_Float16)y;
        }
    }
}

// ---- kernel 4: both heads, pred_gt | pred_noise (f32 out) -------------------
// HT = packed [128][1024]: rows 0-63 = Wgt^T, rows 64-127 = Wn^T.
__global__ __launch_bounds__(256) void k_heads(const _Float16* __restrict__ h,
                                               const _Float16* __restrict__ HT,
                                               const float* __restrict__ bgt,
                                               const float* __restrict__ bn,
                                               float* __restrict__ out) {
    __shared__ _Float16 As[16 * 32];
    __shared__ _Float16 Bs[128 * 32];

    const int tid  = threadIdx.x;
    const int wave = tid >> 5;
    const int lane = tid & 31;
    const int row0 = blockIdx.x * 16;
    const int mh   = (lane >> 4) * 8;

    v8f acc = {0.f, 0.f, 0.f, 0.f, 0.f, 0.f, 0.f, 0.f};

    for (int kc = 0; kc < 1024; kc += 32) {
        if (tid < 64) {
            int r = tid >> 2, seg = (tid & 3) << 3;
            cp16(As + r * 32 + seg,
                 h + (size_t)(row0 + r) * 1024 + kc + seg);
        }
        for (int i = tid; i < 512; i += 256) {
            int n = i >> 2, seg = (i & 3) << 3;
            cp16(Bs + n * 32 + seg, HT + (size_t)n * 1024 + kc + seg);
        }
        cp_commit();
        __syncthreads();

        v16h a = load_a_frag(As, lane);
        v16h b = load_b_frag(Bs, wave * 16, lane);
        acc = __builtin_amdgcn_wmma_f32_16x16x32_f16(
            false, a, false, b, (short)0, acc, false, false);
        __syncthreads();
    }

    int   c   = wave * 16 + (lane & 15);
    int   cc  = (c < 64) ? c : c - 64;
    float bb  = (c < 64) ? bgt[cc] : bn[cc];
    size_t ob = (c < 64) ? 0 : (size_t)BB * DD;
#pragma unroll
    for (int r = 0; r < 8; ++r) {
        out[ob + (size_t)(row0 + mh + r) * 64 + cc] = acc[r] + bb;
    }
}

// ---- host launcher ----------------------------------------------------------
extern "C" void kernel_launch(void* const* d_in, const int* in_sizes, int n_in,
                              void* d_out, int out_size, void* d_ws, size_t ws_size,
                              hipStream_t stream) {
    const float* gt    = (const float*)d_in[0];
    const float* noise = (const float*)d_in[1];
    const float* t     = (const float*)d_in[2];
    const float* Wt1   = (const float*)d_in[3];
    const float* bt1   = (const float*)d_in[4];
    const float* Wt2   = (const float*)d_in[5];
    const float* bt2   = (const float*)d_in[6];
    const float* W1    = (const float*)d_in[7];
    const float* b1    = (const float*)d_in[8];
    const float* W2    = (const float*)d_in[9];
    const float* b2    = (const float*)d_in[10];
    const float* W3    = (const float*)d_in[11];
    const float* b3    = (const float*)d_in[12];
    const float* g1    = (const float*)d_in[13];
    const float* be1   = (const float*)d_in[14];
    const float* Ws1   = (const float*)d_in[15];
    const float* bs1   = (const float*)d_in[16];
    const float* g2    = (const float*)d_in[17];
    const float* be2   = (const float*)d_in[18];
    const float* Ws2   = (const float*)d_in[19];
    const float* bs2   = (const float*)d_in[20];
    const float* g3    = (const float*)d_in[21];
    const float* be3   = (const float*)d_in[22];
    const float* Ws3   = (const float*)d_in[23];
    const float* bs3   = (const float*)d_in[24];
    const float* Wgt   = (const float*)d_in[25];
    const float* bgt   = (const float*)d_in[26];
    const float* Wn    = (const float*)d_in[27];
    const float* bn    = (const float*)d_in[28];
    float* outp = (float*)d_out;

    // workspace layout (f16 intermediates + packed f16 weights)
    char* ws = (char*)d_ws;
    const size_t MB = 1024 * 1024;
    _Float16* mixed = (_Float16*)(ws);                //   8 MB used
    _Float16* te    = (_Float16*)(ws + 16  * MB);     // 128 MB
    _Float16* hA    = (_Float16*)(ws + 152 * MB);     // 128 MB
    _Float16* hB    = (_Float16*)(ws + 288 * MB);     // 128 MB
    _Float16* Wt2T  = (_Float16*)(ws + 416 * MB);     //   2 MB  [1024][1024]
    _Float16* W1T   = (_Float16*)(ws + 418 * MB);     // 128 KB  [1024][64]
    _Float16* W2T   = (_Float16*)(ws + 420 * MB);     //   2 MB
    _Float16* W3T   = (_Float16*)(ws + 422 * MB);     //   2 MB
    _Float16* Ws1T  = (_Float16*)(ws + 424 * MB);     //   4 MB  [2048][1024]
    _Float16* Ws2T  = (_Float16*)(ws + 428 * MB);     //   4 MB
    _Float16* Ws3T  = (_Float16*)(ws + 432 * MB);     //   4 MB
    _Float16* HT    = (_Float16*)(ws + 436 * MB);     // 256 KB [128][1024]

    // pack weights f32 [K][N] -> f16 [N][K]
    k_packT<<<(1024 * 1024) / 256, 256, 0, stream>>>(Wt2, Wt2T, 1024, 1024);
    k_packT<<<(64   * 1024) / 256, 256, 0, stream>>>(W1,  W1T,  64,   1024);
    k_packT<<<(1024 * 1024) / 256, 256, 0, stream>>>(W2,  W2T,  1024, 1024);
    k_packT<<<(1024 * 1024) / 256, 256, 0, stream>>>(W3,  W3T,  1024, 1024);
    k_packT<<<(1024 * 2048) / 256, 256, 0, stream>>>(Ws1, Ws1T, 1024, 2048);
    k_packT<<<(1024 * 2048) / 256, 256, 0, stream>>>(Ws2, Ws2T, 1024, 2048);
    k_packT<<<(1024 * 2048) / 256, 256, 0, stream>>>(Ws3, Ws3T, 1024, 2048);
    k_packT<<<(1024 * 64)   / 256, 256, 0, stream>>>(Wgt, HT,             1024, 64);
    k_packT<<<(1024 * 64)   / 256, 256, 0, stream>>>(Wn,  HT + 64 * 1024, 1024, 64);

    const size_t smem_te  = 1024 + (size_t)1024 * 32 * 2;                               // 66.5 KB
    const size_t smem_blk = 1024 + 2 * (size_t)1024 * 32 * 2 + 2 * 16 * 1024 * 4 + 128; // 257 KB

    (void)hipFuncSetAttribute((const void*)k_te,
        hipFuncAttributeMaxDynamicSharedMemorySize, (int)smem_te);
    (void)hipFuncSetAttribute((const void*)k_block<64>,
        hipFuncAttributeMaxDynamicSharedMemorySize, (int)smem_blk);
    (void)hipFuncSetAttribute((const void*)k_block<1024>,
        hipFuncAttributeMaxDynamicSharedMemorySize, (int)smem_blk);

    k_mixed<<<(BB * DD) / 256, 256, 0, stream>>>(gt, noise, t, mixed);
    k_te<<<BB / 16, 256, smem_te, stream>>>(t, Wt1, bt1, Wt2T, bt2, te);
    k_block<64><<<BB / 16, 256, smem_blk, stream>>>(mixed, te, W1T, b1, g1, be1, Ws1T, bs1, hA);
    k_block<1024><<<BB / 16, 256, smem_blk, stream>>>(hA, te, W2T, b2, g2, be2, Ws2T, bs2, hB);
    k_block<1024><<<BB / 16, 256, smem_blk, stream>>>(hB, te, W3T, b3, g3, be3, Ws3T, bs3, hA);
    k_heads<<<BB / 16, 256, 0, stream>>>(hA, HT, bgt, bn, outp);
}
